// Rope_36438502540111
// MI455X (gfx1250) — matrix operand order repaired and checked by reading, not compiled, not run on hardware
//
#include <hip/hip_runtime.h>

typedef _Float16 half_t;
typedef __attribute__((ext_vector_type(16))) _Float16 v16h;
typedef __attribute__((ext_vector_type(8)))  _Float16 v8h;
typedef __attribute__((ext_vector_type(4)))  _Float16 v4h;
typedef __attribute__((ext_vector_type(8)))  float    v8f;

#define SEQ_LEN 8192
#define D_MODEL 256

// ---------------------------------------------------------------------------
// WMMA fragment helpers (CDNA5 16x16x32 f16 layouts, wave32)
// ---------------------------------------------------------------------------
__device__ __forceinline__ v16h cat8(v8h lo, v8h hi) {
  return __builtin_shufflevector(lo, hi, 0,1,2,3,4,5,6,7,8,9,10,11,12,13,14,15);
}

// A-fragment: 16(M) x 32(K) tile of a row-major [M][K] buffer.
// Lane L<16 : row L,    K = {0..7, 16..23}
// Lane L>=16: row L-16, K = {8..15, 24..31}
__device__ __forceinline__ v16h load_a(const half_t* tile, int stride, int kbase, int lane) {
  const int r = lane & 15, hl = lane >> 4;
  const half_t* p = tile + r * stride + kbase + hl * 8;
  v8h lo = *(const v8h*)p;
  v8h hi = *(const v8h*)(p + 16);
  return cat8(lo, hi);
}

// B-fragment: 32(K) x 16(N); tile points at row n0 of a row-major [N][K] buffer
__device__ __forceinline__ v16h load_b(const half_t* tile, int stride, int kbase, int lane) {
  const int r = lane & 15, hl = lane >> 4;
  const half_t* p = tile + r * stride + kbase + hl * 8;
  v8h lo = *(const v8h*)p;
  v8h hi = *(const v8h*)(p + 16);
  return cat8(lo, hi);
}

__device__ __forceinline__ v8f wmma16(v16h a, v16h b, v8f c) {
  return __builtin_amdgcn_wmma_f32_16x16x32_f16(false, a, false, b, (short)0, c,
                                                false, false);
}

// ---------------------------------------------------------------------------
// CDNA5 async memory->LDS copy (16 bytes per lane, ASYNCcnt-tracked).
// LDS byte address = low 32 bits of the generic pointer (LDS aperture,
// ISA 10.2).
// ---------------------------------------------------------------------------
__device__ __forceinline__ void async_copy16(const half_t* gsrc, half_t* ldst) {
  unsigned l = (unsigned)(unsigned long long)ldst;
  asm volatile("global_load_async_to_lds_b128 %0, %1, off"
               :
               : "v"(l), "v"((unsigned long long)gsrc)
               : "memory");
}

// ---------------------------------------------------------------------------
// Stage 0: f32 -> f16 cast (x and weights). n is always a multiple of 4.
// ---------------------------------------------------------------------------
__global__ void cvt_f32_f16(const float* __restrict__ in, half_t* __restrict__ out, int n) {
  int i = (blockIdx.x * blockDim.x + threadIdx.x) * 4;
  if (i < n) {
    float4 f = *(const float4*)(in + i);
    v4h h;
    h[0] = (half_t)f.x; h[1] = (half_t)f.y; h[2] = (half_t)f.z; h[3] = (half_t)f.w;
    *(v4h*)(out + i) = h;
  }
}

// ---------------------------------------------------------------------------
// Stage 1: Q/K/V projection (x @ W^T + b), RoPE on Q/K, V stored transposed.
// grid = (SEQ/16, 24): y -> matrix m = y/8 (0=Q,1=K,2=V), col block (y%8)*32.
// block = 32 threads (one wave). 16 WMMAs per wave (K = 256).
// ---------------------------------------------------------------------------
__global__ void __launch_bounds__(32, 8)
qkv_rope_kernel(const half_t* __restrict__ xh,
                const half_t* __restrict__ wqh,
                const half_t* __restrict__ wkh,
                const half_t* __restrict__ wvh,
                const float* __restrict__ bq,
                const float* __restrict__ bk,
                const float* __restrict__ bv,
                half_t* __restrict__ qrot,
                half_t* __restrict__ krot,
                half_t* __restrict__ vt) {
  const int lane = threadIdx.x;
  const int r = lane & 15, hl = lane >> 4;
  const int row0 = blockIdx.x * 16;
  const int m    = blockIdx.y >> 3;
  const int col0 = (blockIdx.y & 7) * 32;

  const half_t* W    = (m == 0) ? wqh : (m == 1) ? wkh : wvh;
  const float*  bias = (m == 0) ? bq  : (m == 1) ? bk  : bv;

  v8f acc0 = {}, acc1 = {};
  const half_t* xt = xh + (size_t)row0 * D_MODEL;
#pragma unroll
  for (int c = 0; c < 8; ++c) {
    v16h a  = load_a(xt, D_MODEL, c * 32, lane);
    v16h b0 = load_b(W + (size_t)(col0     ) * D_MODEL, D_MODEL, c * 32, lane);
    v16h b1 = load_b(W + (size_t)(col0 + 16) * D_MODEL, D_MODEL, c * 32, lane);
    acc0 = wmma16(a, b0, acc0);
    acc1 = wmma16(a, b1, acc1);
  }

  const int j0 = col0 + r;
  const int j1 = col0 + 16 + r;
  const float bias0 = bias[j0], bias1 = bias[j1];

  // RoPE constants (reference: freq = 1000^(-(2t)/dim), no position term;
  // rot_odd = x_odd*(sin+cos) replicated faithfully)
  float cs0 = 0.f, sn0 = 0.f, cs1 = 0.f, sn1 = 0.f;
  if (m < 2) {
    float fr0 = powf(1000.0f, -(float)(j0 & ~1) * (1.0f / 256.0f));
    float fr1 = powf(1000.0f, -(float)(j1 & ~1) * (1.0f / 256.0f));
    cs0 = cosf(fr0); sn0 = sinf(fr0);
    cs1 = cosf(fr1); sn1 = sinf(fr1);
  }

#pragma unroll
  for (int v = 0; v < 8; ++v) {
    const int row = row0 + v + hl * 8;   // C/D layout: M = v + 8*(lane/16)
    float y0 = acc0[v] + bias0;
    float y1 = acc1[v] + bias1;
    if (m < 2) {
      float p0 = __shfl_xor(y0, 1, 32);  // partner column (N^1) lives in lane^1
      float p1 = __shfl_xor(y1, 1, 32);
      y0 = (j0 & 1) ? y0 * (sn0 + cs0) : y0 * cs0 - p0 * sn0;
      y1 = (j1 & 1) ? y1 * (sn1 + cs1) : y1 * cs1 - p1 * sn1;
      half_t* dst = (m == 0) ? qrot : krot;
      dst[(size_t)row * D_MODEL + j0] = (half_t)y0;
      dst[(size_t)row * D_MODEL + j1] = (half_t)y1;
    } else {
      vt[(size_t)j0 * SEQ_LEN + row] = (half_t)y0;   // V stored transposed
      vt[(size_t)j1 * SEQ_LEN + row] = (half_t)y1;
    }
  }
}

// ---------------------------------------------------------------------------
// Stage 2: flash attention. block = 64 threads (2 waves), 32 query rows per
// block, key tiles of 32, double-buffered async global->LDS staging.
// Padded LDS strides (264, 40 halves) kill bank conflicts on fragment reads.
// ---------------------------------------------------------------------------
#define QK_STRIDE 264
#define V_STRIDE  40
#define NKT (SEQ_LEN / 32)

// Issue one K/V tile's async copies: 8 (K) + 16 (V) = 24 per thread.
__device__ __forceinline__ void stage_tiles_async(const half_t* __restrict__ kr,
                                                  const half_t* __restrict__ vtg,
                                                  half_t* Kl, half_t* Vl,
                                                  int kt, int tid) {
  {  // K tile: 32 keys x 256 dims
    const int row = tid >> 1;
    const int off = (tid & 1) * 128;
    const half_t* src = kr + (size_t)(kt * 32 + row) * D_MODEL + off;
    half_t* dst = Kl + row * QK_STRIDE + off;
#pragma unroll
    for (int i = 0; i < 8; ++i)
      async_copy16(src + i * 16, dst + i * 16);
  }
  {  // V^T tile: 256 dims x 32 keys
    const int c = (tid & 3) * 8;
    const int rr = tid >> 2;
#pragma unroll
    for (int p = 0; p < 16; ++p) {
      const int drow = p * 16 + rr;
      async_copy16(vtg + (size_t)drow * SEQ_LEN + kt * 32 + c,
                   Vl + drow * V_STRIDE + c);
    }
  }
}

__global__ void __launch_bounds__(64, 1)
attn_kernel(const half_t* __restrict__ qr,
            const half_t* __restrict__ kr,
            const half_t* __restrict__ vtg,
            float* __restrict__ out) {
  __shared__ half_t Qlds[32 * QK_STRIDE];
  __shared__ half_t Klds[2][32 * QK_STRIDE];
  __shared__ half_t Vlds[2][256 * V_STRIDE];
  __shared__ half_t Plds[2][16 * V_STRIDE];

  const int tid  = threadIdx.x;
  const int lane = tid & 31, w = tid >> 5;
  const int r = lane & 15, hl = lane >> 4;
  const int q0 = blockIdx.x * 32;

  // ---- stage this block's Q tile (32 rows x 256) into LDS (one-time)
  {
    const int row = tid >> 1;
    const int off = (tid & 1) * 128;
#pragma unroll
    for (int i = 0; i < 8; ++i)
      *(v8h*)(Qlds + row * QK_STRIDE + off + i * 16) =
          *(const v8h*)(qr + (size_t)(q0 + row) * D_MODEL + off + i * 16);
  }
  // ---- prefetch first K/V tile asynchronously into buffer 0
  stage_tiles_async(kr, vtg, Klds[0], Vlds[0], 0, tid);

  float mrow[8], lrow[8];
  v8f O[16];
  v8f zacc = {};
#pragma unroll
  for (int v = 0; v < 8; ++v) { mrow[v] = -3.0e38f; lrow[v] = 0.0f; }
#pragma unroll
  for (int n = 0; n < 16; ++n) O[n] = zacc;

  const float SCALE = 0.0625f;          // 1/sqrt(256)
  const float LOG2E = 1.44269504089f;
  const half_t* qt = Qlds + w * 16 * QK_STRIDE;
  half_t* pl = Plds[w];

#pragma unroll 1
  for (int kt = 0; kt < NKT; ++kt) {
    const int cur = kt & 1;
    // ---- prefetch next tile into the other buffer, then drain this tile's
    //      batch: each batch is 24 async ops/thread, so waiting ASYNCcnt<=24
    //      guarantees the *previous* batch (tile kt) has landed in LDS.
    if (kt + 1 < NKT) {
      stage_tiles_async(kr, vtg, Klds[cur ^ 1], Vlds[cur ^ 1], kt + 1, tid);
      asm volatile("s_wait_asynccnt 0x18" ::: "memory");
    } else {
      asm volatile("s_wait_asynccnt 0x0" ::: "memory");
    }
    __syncthreads();   // tile kt visible to all waves

    const half_t* Kl = Klds[cur];
    const half_t* Vl = Vlds[cur];

    // ---- S = Q K^T : 16 rows x 32 keys per wave, K-dim = 256 -> 16 WMMAs
    v8f s0 = zacc, s1 = zacc;
#pragma unroll
    for (int c = 0; c < 8; ++c) {
      v16h a  = load_a(qt, QK_STRIDE, c * 32, lane);
      v16h b0 = load_b(Kl,                  QK_STRIDE, c * 32, lane);
      v16h b1 = load_b(Kl + 16 * QK_STRIDE, QK_STRIDE, c * 32, lane);
      s0 = wmma16(a, b0, s0);
      s1 = wmma16(a, b1, s1);
    }

    // ---- online softmax (rows live per accumulator index; 16-lane reductions)
#pragma unroll
    for (int v = 0; v < 8; ++v) {
      float t = fmaxf(s0[v], s1[v]);
      t = fmaxf(t, __shfl_xor(t, 1, 32));
      t = fmaxf(t, __shfl_xor(t, 2, 32));
      t = fmaxf(t, __shfl_xor(t, 4, 32));
      t = fmaxf(t, __shfl_xor(t, 8, 32));
      const float mnew = fmaxf(mrow[v], t * SCALE);
      const float resc = exp2f((mrow[v] - mnew) * LOG2E);
      const float p0 = exp2f((s0[v] * SCALE - mnew) * LOG2E);
      const float p1 = exp2f((s1[v] * SCALE - mnew) * LOG2E);
      float rs = p0 + p1;
      rs += __shfl_xor(rs, 1, 32);
      rs += __shfl_xor(rs, 2, 32);
      rs += __shfl_xor(rs, 4, 32);
      rs += __shfl_xor(rs, 8, 32);
      lrow[v] = lrow[v] * resc + rs;
      mrow[v] = mnew;
#pragma unroll
      for (int n = 0; n < 16; ++n) O[n][v] *= resc;
      // C-layout -> LDS (per-wave region) for A-layout reload
      pl[(v + hl * 8) * V_STRIDE + r]      = (half_t)p0;
      pl[(v + hl * 8) * V_STRIDE + r + 16] = (half_t)p1;
    }

    // ---- O += P @ V : 16 dim-tiles -> 16 WMMAs
    v16h pa = load_a(pl, V_STRIDE, 0, lane);
#pragma unroll
    for (int n = 0; n < 16; ++n) {
      v16h bv = load_b(Vl + n * 16 * V_STRIDE, V_STRIDE, 0, lane);
      O[n] = wmma16(pa, bv, O[n]);
    }

    __syncthreads();   // all waves done with buffer `cur` before it is refilled
  }

  // ---- normalize and write (f32)
  float invl[8];
#pragma unroll
  for (int v = 0; v < 8; ++v) invl[v] = 1.0f / lrow[v];
  const int row = q0 + w * 16;
#pragma unroll
  for (int n = 0; n < 16; ++n) {
#pragma unroll
    for (int v = 0; v < 8; ++v)
      out[(size_t)(row + v + hl * 8) * D_MODEL + n * 16 + r] = O[n][v] * invl[v];
  }
}

// ---------------------------------------------------------------------------
// Host-side launch. Workspace: xh/qrot/krot/vt (4MB each) + 3x128KB weights.
// ---------------------------------------------------------------------------
extern "C" void kernel_launch(void* const* d_in, const int* in_sizes, int n_in,
                              void* d_out, int out_size, void* d_ws, size_t ws_size,
                              hipStream_t stream) {
  const float* x  = (const float*)d_in[0];
  const float* wq = (const float*)d_in[1];
  const float* bq = (const float*)d_in[2];
  const float* wk = (const float*)d_in[3];
  const float* bk = (const float*)d_in[4];
  const float* wv = (const float*)d_in[5];
  const float* bv = (const float*)d_in[6];

  char* ws = (char*)d_ws;
  half_t* xh   = (half_t*)ws; ws += (size_t)SEQ_LEN * D_MODEL * sizeof(half_t);
  half_t* qrot = (half_t*)ws; ws += (size_t)SEQ_LEN * D_MODEL * sizeof(half_t);
  half_t* krot = (half_t*)ws; ws += (size_t)SEQ_LEN * D_MODEL * sizeof(half_t);
  half_t* vt   = (half_t*)ws; ws += (size_t)SEQ_LEN * D_MODEL * sizeof(half_t);
  half_t* wqh  = (half_t*)ws; ws += (size_t)D_MODEL * D_MODEL * sizeof(half_t);
  half_t* wkh  = (half_t*)ws; ws += (size_t)D_MODEL * D_MODEL * sizeof(half_t);
  half_t* wvh  = (half_t*)ws; ws += (size_t)D_MODEL * D_MODEL * sizeof(half_t);

  const int nx = SEQ_LEN * D_MODEL;
  const int nw = D_MODEL * D_MODEL;
  cvt_f32_f16<<<(nx / 4 + 255) / 256, 256, 0, stream>>>(x,  xh,  nx);
  cvt_f32_f16<<<(nw / 4 + 255) / 256, 256, 0, stream>>>(wq, wqh, nw);
  cvt_f32_f16<<<(nw / 4 + 255) / 256, 256, 0, stream>>>(wk, wkh, nw);
  cvt_f32_f16<<<(nw / 4 + 255) / 256, 256, 0, stream>>>(wv, wvh, nw);

  qkv_rope_kernel<<<dim3(SEQ_LEN / 16, 24), 32, 0, stream>>>(
      xh, wqh, wkh, wvh, bq, bk, bv, qrot, krot, vt);

  attn_kernel<<<SEQ_LEN / 32, 64, 0, stream>>>(qrot, krot, vt, (float*)d_out);
}
